// QAconcept_89730456748857
// MI455X (gfx1250) — compile-verified, matrix operand order checked by reading, not run on hardware
//
#include <hip/hip_runtime.h>
#include <hip/hip_bf16.h>
#include <cstddef>

// ---------------------------------------------------------------------------
// Types for CDNA5 WMMA (wave32)
// ---------------------------------------------------------------------------
typedef _Float16 v16h  __attribute__((ext_vector_type(16)));
typedef _Float16 half8 __attribute__((ext_vector_type(8)));
typedef float    v8f   __attribute__((ext_vector_type(8)));

union FragU { v16h v; half8 h[2]; };

// Load one 16-half A/B fragment for v_wmma_f32_16x16x32_f16.
// Caller passes &buf[row*RS + k0 + hs*8]; halfs 0-7 = K[k0+hs*8 ..+8),
// halfs 8-15 = K[k0+16+hs*8 ..+8)  (matches ISA 16-bit 16x32 layout).
__device__ __forceinline__ v16h load_frag(const _Float16* base) {
  FragU f;
  f.h[0] = *(const half8*)(base);
  f.h[1] = *(const half8*)(base + 16);
  return f.v;
}

static constexpr int Bb = 4, T = 128, D = 512, Hh = 4, NL = 4, DFF = 2048, NC = 5;
static constexpr int M_TOK = Bb * T;  // 512 rows (tokens) per encoder state

// ---------------------------------------------------------------------------
// WMMA GEMM: C[M,N] = op(A[M,K] @ Bw[K,N] + bias[N]) (+ residual)
// 128x128 C tile per 256-thread block; wave w owns rows [w*16, w*16+16).
// A is fp32 (activations), Bw is fp32 weights [K,N] row-major; both converted
// to f16 while staging into LDS (B stored transposed [n][k] in LDS).
// ---------------------------------------------------------------------------
template<bool GELU, bool RES>
__global__ __launch_bounds__(256)
void gemm_wmma(const float* __restrict__ A, const float* __restrict__ Bw,
               const float* __restrict__ bias, const float* __restrict__ Res,
               float* __restrict__ C, int N, int K)
{
  __shared__ _Float16 sA[128 * 32];   // [m][k]
  __shared__ _Float16 sB[128 * 32];   // [n][k]
  const int tid  = threadIdx.x;
  const int m0   = blockIdx.y * 128;
  const int n0   = blockIdx.x * 128;
  const int wid  = tid >> 5;
  const int lane = tid & 31;
  const int ml   = lane & 15;
  const int hs   = lane >> 4;

  const v8f vzero = {0.f, 0.f, 0.f, 0.f, 0.f, 0.f, 0.f, 0.f};
  v8f acc[8];
#pragma unroll
  for (int j = 0; j < 8; ++j) acc[j] = vzero;

  const int arow = tid >> 1;          // 0..127
  const int acb  = (tid & 1) * 16;    // 0 / 16
  const int bkk  = tid >> 3;          // 0..31
  const int bnb  = (tid & 7) * 16;    // 0..112

  for (int k0 = 0; k0 < K; k0 += 32) {
    // stage A tile (fp32 -> f16)
    {
      const float* as = A + (size_t)(m0 + arow) * K + k0 + acb;
      if (k0 + 32 < K) __builtin_prefetch(as + 32, 0, 1);  // global_prefetch_b8
      _Float16* dst = &sA[arow * 32 + acb];
#pragma unroll
      for (int i = 0; i < 16; i += 4) {
        float4 f = *(const float4*)(as + i);
        dst[i + 0] = (_Float16)f.x; dst[i + 1] = (_Float16)f.y;
        dst[i + 2] = (_Float16)f.z; dst[i + 3] = (_Float16)f.w;
      }
    }
    // stage B tile transposed (W[k][n] -> sB[n][k], fp32 -> f16)
    {
      const float* bs = Bw + (size_t)(k0 + bkk) * N + n0 + bnb;
#pragma unroll
      for (int i = 0; i < 16; i += 4) {
        float4 f = *(const float4*)(bs + i);
        sB[(bnb + i + 0) * 32 + bkk] = (_Float16)f.x;
        sB[(bnb + i + 1) * 32 + bkk] = (_Float16)f.y;
        sB[(bnb + i + 2) * 32 + bkk] = (_Float16)f.z;
        sB[(bnb + i + 3) * 32 + bkk] = (_Float16)f.w;
      }
    }
    __syncthreads();

    v16h a = load_frag(&sA[(wid * 16 + ml) * 32 + hs * 8]);
#pragma unroll
    for (int j = 0; j < 8; ++j) {
      v16h b = load_frag(&sB[(j * 16 + ml) * 32 + hs * 8]);
      acc[j] = __builtin_amdgcn_wmma_f32_16x16x32_f16(
          false, a, false, b, (short)0, acc[j], false, false);
    }
    __syncthreads();
  }

  // epilogue: C layout lane(ml)=col, hs*8+r=row-in-tile
#pragma unroll
  for (int j = 0; j < 8; ++j) {
    const int col = n0 + j * 16 + ml;
    const float bv = bias[col];
#pragma unroll
    for (int r = 0; r < 8; ++r) {
      const int row = m0 + wid * 16 + hs * 8 + r;
      float v = acc[j][r] + bv;
      if (GELU) v = v * 0.5f * (1.0f + erff(v * 0.70710678118654752f));
      if (RES)  v += Res[(size_t)row * N + col];
      C[(size_t)row * N + col] = v;
    }
  }
}

// ---------------------------------------------------------------------------
// Fused per-(b,h) attention: S = QK^T/sqrt(dk), softmax, O = P V, all WMMA.
// One 256-thread block per (b,h). dk = 128, T = 128.
// ---------------------------------------------------------------------------
__global__ __launch_bounds__(256)
void attn_wmma(const float* __restrict__ Q, const float* __restrict__ Kt,
               const float* __restrict__ V, float* __restrict__ O)
{
  __shared__ _Float16 s1[128 * 128];  // Q (f16), then P (f16)
  __shared__ _Float16 s2[128 * 128];  // K rows (f16), then V^T (f16)
  const int tid  = threadIdx.x;
  const int bi   = blockIdx.x >> 2;
  const int hh   = blockIdx.x & 3;
  const int wid  = tid >> 5, lane = tid & 31, ml = lane & 15, hs = lane >> 4;
  const size_t rowbase = (size_t)(bi * 128) * D + hh * 128;

  // stage Q and K (row-major) as f16
  {
    const int row = tid >> 1, cb = (tid & 1) * 64;
    const float* qs = Q + rowbase + (size_t)row * D + cb;
    const float* ks = Kt + rowbase + (size_t)row * D + cb;
#pragma unroll 4
    for (int i = 0; i < 64; i += 4) {
      float4 f = *(const float4*)(qs + i);
      s1[row * 128 + cb + i + 0] = (_Float16)f.x; s1[row * 128 + cb + i + 1] = (_Float16)f.y;
      s1[row * 128 + cb + i + 2] = (_Float16)f.z; s1[row * 128 + cb + i + 3] = (_Float16)f.w;
      float4 g = *(const float4*)(ks + i);
      s2[row * 128 + cb + i + 0] = (_Float16)g.x; s2[row * 128 + cb + i + 1] = (_Float16)g.y;
      s2[row * 128 + cb + i + 2] = (_Float16)g.z; s2[row * 128 + cb + i + 3] = (_Float16)g.w;
    }
  }
  __syncthreads();

  const v8f vzero = {0.f, 0.f, 0.f, 0.f, 0.f, 0.f, 0.f, 0.f};
  v8f acc[8];
#pragma unroll
  for (int j = 0; j < 8; ++j) acc[j] = vzero;

  // S = Q @ K^T   (B-layout [n=key_token][k=d] == K row-major)
#pragma unroll
  for (int k0 = 0; k0 < 128; k0 += 32) {
    v16h a = load_frag(&s1[(wid * 16 + ml) * 128 + k0 + hs * 8]);
#pragma unroll
    for (int j = 0; j < 8; ++j) {
      v16h b = load_frag(&s2[(j * 16 + ml) * 128 + k0 + hs * 8]);
      acc[j] = __builtin_amdgcn_wmma_f32_16x16x32_f16(
          false, a, false, b, (short)0, acc[j], false, false);
    }
  }
  __syncthreads();  // everyone done reading s1(Q)/s2(K)

  // row softmax: row data for (hs, r) is spread across lanes of same half-wave
  const float scale = 0.08838834764831845f;  // 1/sqrt(128)
#pragma unroll
  for (int r = 0; r < 8; ++r) {
    float sv[8];
    float mx = -3.4e38f;
#pragma unroll
    for (int j = 0; j < 8; ++j) { sv[j] = acc[j][r] * scale; mx = fmaxf(mx, sv[j]); }
#pragma unroll
    for (int m = 1; m < 16; m <<= 1) mx = fmaxf(mx, __shfl_xor(mx, m, 32));
    float sum = 0.f;
#pragma unroll
    for (int j = 0; j < 8; ++j) { sv[j] = __expf(sv[j] - mx); sum += sv[j]; }
#pragma unroll
    for (int m = 1; m < 16; m <<= 1) sum += __shfl_xor(sum, m, 32);
    const float inv = 1.0f / sum;
    const int row = wid * 16 + hs * 8 + r;
#pragma unroll
    for (int j = 0; j < 8; ++j)
      s1[row * 128 + j * 16 + ml] = (_Float16)(sv[j] * inv);  // P (f16)
  }

  // stage V^T: s2[d][tok]  (B-layout [n=d][k=tok])
  {
    const int tok = tid & 127, dbase = (tid >> 7) * 64;
    const float* vs = V + rowbase + (size_t)tok * D + dbase;
#pragma unroll 4
    for (int i = 0; i < 64; ++i)
      s2[(dbase + i) * 128 + tok] = (_Float16)vs[i];
  }
  __syncthreads();

#pragma unroll
  for (int j = 0; j < 8; ++j) acc[j] = vzero;

  // O = P @ V
#pragma unroll
  for (int k0 = 0; k0 < 128; k0 += 32) {
    v16h a = load_frag(&s1[(wid * 16 + ml) * 128 + k0 + hs * 8]);
#pragma unroll
    for (int j = 0; j < 8; ++j) {
      v16h b = load_frag(&s2[(j * 16 + ml) * 128 + k0 + hs * 8]);
      acc[j] = __builtin_amdgcn_wmma_f32_16x16x32_f16(
          false, a, false, b, (short)0, acc[j], false, false);
    }
  }

#pragma unroll
  for (int j = 0; j < 8; ++j) {
    const int col = j * 16 + ml;
#pragma unroll
    for (int r = 0; r < 8; ++r) {
      const int row = wid * 16 + hs * 8 + r;
      O[rowbase + (size_t)row * D + col] = acc[j][r];
    }
  }
}

// ---------------------------------------------------------------------------
// LayerNorm over D=512 per row; ADDPOS=1 adds pos_emb[row % 128] first.
// One 128-thread block per row.
// ---------------------------------------------------------------------------
template<int ADDPOS>
__global__ __launch_bounds__(128)
void ln_kernel(const float* __restrict__ src, const float* __restrict__ pos,
               const float* __restrict__ w, const float* __restrict__ b,
               float* __restrict__ dst)
{
  __shared__ float red[128];
  const int row = blockIdx.x, tid = threadIdx.x;
  float v[4];
#pragma unroll
  for (int k = 0; k < 4; ++k) {
    const int c = tid + k * 128;
    float x = src[(size_t)row * D + c];
    if (ADDPOS) x += pos[(size_t)(row & 127) * D + c];
    v[k] = x;
  }
  float s = v[0] + v[1] + v[2] + v[3];
  red[tid] = s; __syncthreads();
  for (int o = 64; o > 0; o >>= 1) { if (tid < o) red[tid] += red[tid + o]; __syncthreads(); }
  const float mean = red[0] * (1.0f / D); __syncthreads();
  float q = 0.f;
#pragma unroll
  for (int k = 0; k < 4; ++k) { const float d = v[k] - mean; q += d * d; }
  red[tid] = q; __syncthreads();
  for (int o = 64; o > 0; o >>= 1) { if (tid < o) red[tid] += red[tid + o]; __syncthreads(); }
  const float inv = rsqrtf(red[0] * (1.0f / D) + 1e-12f); __syncthreads();
#pragma unroll
  for (int k = 0; k < 4; ++k) {
    const int c = tid + k * 128;
    dst[(size_t)row * D + c] = w[c] * (v[k] - mean) * inv + b[c];
  }
}

// ---------------------------------------------------------------------------
// Relation similarity attention head (NC=5 inner dim -> VALU).
// One 128-thread block per (b, t1); thread s handles key index s.
// ---------------------------------------------------------------------------
__device__ __forceinline__ float blk_max(float v, float* red, int tid) {
  red[tid] = v; __syncthreads();
  for (int o = 64; o > 0; o >>= 1) { if (tid < o) red[tid] = fmaxf(red[tid], red[tid + o]); __syncthreads(); }
  const float r = red[0]; __syncthreads(); return r;
}
__device__ __forceinline__ float blk_sum(float v, float* red, int tid) {
  red[tid] = v; __syncthreads();
  for (int o = 64; o > 0; o >>= 1) { if (tid < o) red[tid] += red[tid + o]; __syncthreads(); }
  const float r = red[0]; __syncthreads(); return r;
}

__global__ __launch_bounds__(128)
void simattn_kernel(const float* __restrict__ q, const float* __restrict__ k,
                    const float* __restrict__ qa, const float* __restrict__ aq,
                    const float* __restrict__ E, float* __restrict__ xout)
{
  __shared__ float sE[NC * D];
  __shared__ float sQ[D];
  __shared__ float sP[128];
  __shared__ float sRR[128 * NC];
  __shared__ float red[128];
  const int bi = blockIdx.x >> 7, t1 = blockIdx.x & 127, s = threadIdx.x;

  for (int i = s; i < NC * D; i += 128) sE[i] = E[i];
  for (int i = s; i < D; i += 128) sQ[i] = q[(size_t)(bi * 128 + t1) * D + i];

  float rL[NC], rR[NC];
#pragma unroll
  for (int c = 0; c < NC; ++c)
    rL[c] = qa[(((size_t)bi * T + t1) * T + s) * NC + c];
#pragma unroll
  for (int c = 0; c < NC; ++c) {
    rR[c] = aq[(((size_t)bi * T + s) * T + t1) * NC + c];
    sRR[s * NC + c] = rR[c];
  }
  __syncthreads();

  const float* krow = k + (size_t)(bi * 128 + s) * D;
  float score = 0.f;
  for (int d = 0; d < D; ++d) {
    const float e0 = sE[d], e1 = sE[D + d], e2 = sE[2 * D + d],
                e3 = sE[3 * D + d], e4 = sE[4 * D + d];
    const float hl = rL[0]*e0 + rL[1]*e1 + rL[2]*e2 + rL[3]*e3 + rL[4]*e4;
    const float hr = rR[0]*e0 + rR[1]*e1 + rR[2]*e2 + rR[3]*e3 + rR[4]*e4;
    score += (sQ[d] + hl) * (krow[d] + hr);
  }

  // softmax over s
  float mx = blk_max(score, red, s);
  float p  = __expf(score - mx);
  p /= blk_sum(p, red, s);
  // sharpening: softmax(1000*p), clipped
  const float z = 1000.f * p;
  const float mz = blk_max(z, red, s);
  float p2 = __expf(z - mz);
  p2 /= blk_sum(p2, red, s);
  p2 = fminf(fmaxf(p2, 0.f), 1.f);
  sP[s] = p2;
  __syncthreads();

  // x[b,t1,d] = sum_s p2[s] * (k[b,s,d] + hR[b,s,t1,d])
#pragma unroll
  for (int kk = 0; kk < 4; ++kk) {
    const int d = s + kk * 128;
    const float e0 = sE[d], e1 = sE[D + d], e2 = sE[2 * D + d],
                e3 = sE[3 * D + d], e4 = sE[4 * D + d];
    float x = 0.f;
    for (int ss = 0; ss < 128; ++ss) {
      const float* rr = &sRR[ss * NC];
      const float hr = rr[0]*e0 + rr[1]*e1 + rr[2]*e2 + rr[3]*e3 + rr[4]*e4;
      x += sP[ss] * (k[(size_t)(bi * 128 + ss) * D + d] + hr);
    }
    xout[(size_t)(bi * 128 + t1) * D + d] = x;
  }
}

// ---------------------------------------------------------------------------
// out[b, 0..2] = mean_t(x[b,t,:]) @ cls_w + cls_b   (one block per batch)
// ---------------------------------------------------------------------------
__global__ __launch_bounds__(256)
void cls_kernel(const float* __restrict__ x, const float* __restrict__ w,
                const float* __restrict__ bb, float* __restrict__ out)
{
  __shared__ float red[256];
  const int bi = blockIdx.x, tid = threadIdx.x;
  float p[3] = {0.f, 0.f, 0.f};
  for (int d = tid; d < D; d += 256) {
    float m = 0.f;
    for (int t = 0; t < T; ++t) m += x[(size_t)(bi * T + t) * D + d];
    m *= (1.0f / T);
#pragma unroll
    for (int j = 0; j < 3; ++j) p[j] += m * w[d * 3 + j];
  }
#pragma unroll
  for (int j = 0; j < 3; ++j) {
    red[tid] = p[j]; __syncthreads();
    for (int o = 128; o > 0; o >>= 1) { if (tid < o) red[tid] += red[tid + o]; __syncthreads(); }
    if (tid == 0) out[bi * 3 + j] = red[0] + bb[j];
    __syncthreads();
  }
}

// ---------------------------------------------------------------------------
// Host orchestration
// ---------------------------------------------------------------------------
extern "C" void kernel_launch(void* const* d_in, const int* in_sizes, int n_in,
                              void* d_out, int out_size, void* d_ws, size_t ws_size,
                              hipStream_t stream) {
  (void)in_sizes; (void)n_in; (void)out_size; (void)ws_size;
  const float* q_emb    = (const float*)d_in[0];
  const float* a_emb    = (const float*)d_in[1];
  const float* qa_rel   = (const float*)d_in[2];
  const float* aq_rel   = (const float*)d_in[3];
  const float* concept  = (const float*)d_in[4];
  const float* pos_emb  = (const float*)d_in[5];
  const float* pe_w     = (const float*)d_in[6];
  const float* pe_b     = (const float*)d_in[7];
  const float* Wq       = (const float*)d_in[8];
  const float* bq       = (const float*)d_in[9];
  const float* Wk       = (const float*)d_in[10];
  const float* bk       = (const float*)d_in[11];
  const float* Wv       = (const float*)d_in[12];
  const float* bv       = (const float*)d_in[13];
  const float* Wo       = (const float*)d_in[14];
  const float* bo       = (const float*)d_in[15];
  const float* ff1_w    = (const float*)d_in[16];
  const float* ff1_b    = (const float*)d_in[17];
  const float* ff2_w    = (const float*)d_in[18];
  const float* ff2_b    = (const float*)d_in[19];
  const float* lnin_w   = (const float*)d_in[20];
  const float* lnin_b   = (const float*)d_in[21];
  const float* lnout_w  = (const float*)d_in[22];
  const float* lnout_b  = (const float*)d_in[23];
  const float* sim_Wq   = (const float*)d_in[24];
  const float* sim_bq   = (const float*)d_in[25];
  const float* sim_Wk   = (const float*)d_in[26];
  const float* sim_bk   = (const float*)d_in[27];
  const float* cls_w    = (const float*)d_in[28];
  const float* cls_b    = (const float*)d_in[29];
  float* out = (float*)d_out;

  const size_t SZ = (size_t)M_TOK * D;   // 512*512
  float* ws    = (float*)d_ws;
  float* xq    = ws;               // [512,512]
  float* xa    = xq   + SZ;        // [512,512]
  float* qb    = xa   + SZ;        // [512,512]
  float* kb    = qb   + SZ;        // [512,512]
  float* vb    = kb   + SZ;        // [512,512]
  float* attnb = vb   + SZ;        // [512,512] (later reused as sim-attn output)
  float* tmpb  = attnb + SZ;       // [512,512]
  float* ffh   = tmpb + SZ;        // [512,2048]

  const dim3 blk(256);
  const dim3 gD (D / 128,   M_TOK / 128);  // N=512 GEMMs
  const dim3 gF (DFF / 128, M_TOK / 128);  // N=2048 GEMM

  auto run_encoder = [&](const float* emb, float* x) {
    // x = LN(emb + pos_emb[:T]) with pe_w/pe_b
    ln_kernel<1><<<M_TOK, 128, 0, stream>>>(emb, pos_emb, pe_w, pe_b, x);
    for (int l = 0; l < NL; ++l) {
      const size_t wo = (size_t)l * D * D;
      // Q/K/V projections
      gemm_wmma<false, false><<<gD, blk, 0, stream>>>(x, Wq + wo, bq + l * D, nullptr, qb, D, D);
      gemm_wmma<false, false><<<gD, blk, 0, stream>>>(x, Wk + wo, bk + l * D, nullptr, kb, D, D);
      gemm_wmma<false, false><<<gD, blk, 0, stream>>>(x, Wv + wo, bv + l * D, nullptr, vb, D, D);
      // fused attention per (b,h)
      attn_wmma<<<Bb * Hh, blk, 0, stream>>>(qb, kb, vb, attnb);
      // x = attn @ Wo + bo + x
      gemm_wmma<false, true><<<gD, blk, 0, stream>>>(attnb, Wo + wo, bo + l * D, x, x, D, D);
      // h = LN_in(x)
      ln_kernel<0><<<M_TOK, 128, 0, stream>>>(x, nullptr, lnin_w + l * D, lnin_b + l * D, tmpb);
      // ffh = gelu(h @ ff1 + b1)
      gemm_wmma<true, false><<<gF, blk, 0, stream>>>(tmpb, ff1_w + (size_t)l * D * DFF,
                                                     ff1_b + l * DFF, nullptr, ffh, DFF, D);
      // x = ffh @ ff2 + b2 + x
      gemm_wmma<false, true><<<gD, blk, 0, stream>>>(ffh, ff2_w + (size_t)l * DFF * D,
                                                     ff2_b + l * D, x, x, D, DFF);
      // x = LN_out(x) (in place)
      ln_kernel<0><<<M_TOK, 128, 0, stream>>>(x, nullptr, lnout_w + l * D, lnout_b + l * D, x);
    }
  };

  run_encoder(q_emb, xq);
  run_encoder(a_emb, xa);

  // similarity projections (reuse qb/kb)
  gemm_wmma<false, false><<<gD, blk, 0, stream>>>(xq, sim_Wq, sim_bq, nullptr, qb, D, D);
  gemm_wmma<false, false><<<gD, blk, 0, stream>>>(xa, sim_Wk, sim_bk, nullptr, kb, D, D);

  // relation attention head -> attnb reused as x [B,T1,D]
  simattn_kernel<<<Bb * T, 128, 0, stream>>>(qb, kb, qa_rel, aq_rel, concept, attnb);

  // mean over T1 and classifier
  cls_kernel<<<Bb, 256, 0, stream>>>(attnb, cls_w, cls_b, out);
}
